// GraphAttentionLayer1_60662118088928
// MI455X (gfx1250) — matrix operand order repaired and checked by reading, not compiled, not run on hardware
//
#include <hip/hip_runtime.h>

// ---------------------------------------------------------------------------
// GAT layer for MI455X (gfx1250), wave32 + WMMA 16x16x32 bf16.
//   B=8, H=8, N=1024, Fin=256, Fout=128
//   out[b,n,o] = sum_h softmax_m( (q q^T)*scale*adj )[n,m] * q[h,b,m,o]
// ---------------------------------------------------------------------------

#define HN   8
#define BN   8
#define NN   1024
#define FIN  256
#define FOUT 128

typedef __attribute__((ext_vector_type(8)))  float   v8f;
typedef __attribute__((ext_vector_type(4)))  __bf16  v4bf;
typedef __attribute__((ext_vector_type(8)))  __bf16  v8bf;
typedef __attribute__((ext_vector_type(16))) __bf16  v16bf;

#define WMMA_BF16(A, B, C) \
  __builtin_amdgcn_wmma_f32_16x16x32_bf16(false, (A), false, (B), (short)0, (C), false, false)

// ---------------------------------------------------------------------------
// Kernel 0: one-shot f32 -> bf16 conversion (memory bound, ~12MB traffic).
//   blocks [0,2048):  X [B,N,Fin]  -> Xbf  (same layout), 4 elems/thread
//   blocks [2048,3072): W [H,Fin,Fout] -> Wt [H,Fout,Fin] (transposed)
// ---------------------------------------------------------------------------
__global__ __launch_bounds__(256)
void gat_cvt_kernel(const float* __restrict__ X, const float* __restrict__ W,
                    __bf16* __restrict__ Xbf, __bf16* __restrict__ Wt) {
  const int bid = blockIdx.x;
  if (bid < 2048) {
    size_t i = (size_t)bid * 256 + threadIdx.x;        // float4 index
    float4 v = ((const float4*)X)[i];
    v4bf o;
    o[0] = (__bf16)v.x; o[1] = (__bf16)v.y; o[2] = (__bf16)v.z; o[3] = (__bf16)v.w;
    ((v4bf*)Xbf)[i] = o;
  } else {
    size_t i = (size_t)(bid - 2048) * 256 + threadIdx.x;  // element index in Wt
    size_t k = i % FIN;
    size_t t = i / FIN;
    size_t f = t % FOUT;
    size_t h = t / FOUT;
    Wt[i] = (__bf16)W[(h * FIN + k) * FOUT + f];          // coalesced store
  }
}

// ---------------------------------------------------------------------------
// Kernel 1: q[h,b] = X[b] @ W[h]  -> Qb (bf16 row-major [h,b,n,f])
//                                    Qt (bf16 transposed [h,b,f,n])
// grid (H*B, N/128), block 256 (8 waves). wave w owns fout cols [16w,16w+16).
// Pure bf16 loads + WMMA in the hot loop (no conversion VALU).
// ---------------------------------------------------------------------------
__global__ __launch_bounds__(256)
void gat_proj_kernel(const __bf16* __restrict__ Xbf, const __bf16* __restrict__ Wt,
                     __bf16* __restrict__ Qb, __bf16* __restrict__ Qt) {
  const int hb   = blockIdx.x;             // h*BN + b
  const int h    = hb / BN;
  const int b    = hb % BN;
  const int nblk = blockIdx.y * 128;
  const int wave = threadIdx.x >> 5;
  const int lane = threadIdx.x & 31;
  const int hf   = lane >> 4;              // half-wave: 0 or 1
  const int col  = lane & 15;
  const int f0   = wave * 16;

  const __bf16* Wth = Wt + (size_t)h * FOUT * FIN;
  const __bf16* Xbb = Xbf + (size_t)b * NN * FIN;
  __bf16* Qbh = Qb + (size_t)hb * NN * FOUT;
  __bf16* Qth = Qt + (size_t)hb * FOUT * NN;

  // B-fragments of W[h][:, f0:f0+16]: element j -> K = kc*32 + hf*16 + j,
  // col = lane%16  ==  Wt[f0+col][kc*32 + hf*16 + j]  (one 32B load each).
  v16bf wB[8];
#pragma unroll
  for (int kc = 0; kc < 8; ++kc)
    wB[kc] = *(const v16bf*)(Wth + (size_t)(f0 + col) * FIN + kc * 32 + hf * 16);

  for (int rt = 0; rt < 8; ++rt) {
    const int n0 = nblk + rt * 16;
    v8f acc = {};
#pragma unroll
    for (int kc = 0; kc < 8; ++kc) {
      // A layout: row = lane%16; element j -> K = kc*32+hf*16+j.
      const __bf16* xr = Xbb + (size_t)(n0 + col) * FIN + kc * 32 + hf * 16;
      v8bf lo = *(const v8bf*)(xr);
      v8bf hi = *(const v8bf*)(xr + 8);
      v16bf aX = __builtin_shufflevector(lo, hi, 0,1,2,3,4,5,6,7,8,9,10,11,12,13,14,15);
      acc = WMMA_BF16(aX, wB[kc], acc);
    }
    v8bf qv = __builtin_convertvector(acc, v8bf);   // v_cvt_pk_bf16_f32 path
    // C layout: element r -> row n0 + r + 8*hf, col f0 + lane%16.
#pragma unroll
    for (int r = 0; r < 8; ++r)
      Qbh[(size_t)(n0 + r + 8 * hf) * FOUT + f0 + col] = qv[r];
    // Transposed store: fixed fout col, 8 consecutive n -> one 16B store.
    *(v8bf*)(Qth + (size_t)(f0 + col) * NN + n0 + 8 * hf) = qv;
  }
}

// ---------------------------------------------------------------------------
// Kernel 2: flash-attention over m with multiplicative adj, head-sum in LDS.
// grid (B, N/16), block 256; wave = head. Each wave: 16 query rows x all keys.
// ---------------------------------------------------------------------------
__global__ __launch_bounds__(256)
void gat_attn_kernel(const float* __restrict__ adj,
                     const __bf16* __restrict__ Qb,
                     const __bf16* __restrict__ Qt,
                     float* __restrict__ out) {
  const int b    = blockIdx.x;
  const int n0   = blockIdx.y * 16;
  const int h    = threadIdx.x >> 5;       // wave == head
  const int lane = threadIdx.x & 31;
  const int hf   = lane >> 4;
  const int col  = lane & 15;

  __shared__ float adjtile[16][32];                 //  2 KB, shared by all heads
  __shared__ alignas(16) __bf16 pbuf[8][16 * 40];   // 10 KB, per-wave P transpose
  __shared__ float oredux[8][16][17];               // 8.5 KB, head-sum staging

  const int hb = h * BN + b;
  const __bf16* Qbh  = Qb + (size_t)hb * NN * FOUT;
  const __bf16* Qth  = Qt + (size_t)hb * FOUT * NN;
  const float*  adjb = adj + ((size_t)b * NN + n0) * NN;

  // Q-row A-fragments (reused all iterations). K-dim = Fout = 128 -> 4 chunks.
  v16bf aQ[4];
#pragma unroll
  for (int kc = 0; kc < 4; ++kc) {
    const __bf16* qr = Qbh + (size_t)(n0 + col) * FOUT + kc * 32 + hf * 16;
    v8bf lo = *(const v8bf*)(qr);
    v8bf hi = *(const v8bf*)(qr + 8);
    aQ[kc] = __builtin_shufflevector(lo, hi, 0,1,2,3,4,5,6,7,8,9,10,11,12,13,14,15);
  }

  v8f O[8];
#pragma unroll
  for (int c = 0; c < 8; ++c) O[c] = (v8f){};
  float mrow[8], lrow[8];
#pragma unroll
  for (int r = 0; r < 8; ++r) { mrow[r] = -3.0e38f; lrow[r] = 0.0f; }
  const float scale = 0.088388347648318447f;   // 1/sqrt(128)
  __bf16* pb = &pbuf[h][0];

  for (int m0 = 0; m0 < NN; m0 += 32) {
    // ---- stage adj tile [16 x 32] cooperatively (one copy for all 8 heads)
    {
      int e  = threadIdx.x * 2;
      int rr = e >> 5, cc = e & 31;
      float2 v = *(const float2*)(adjb + (size_t)rr * NN + m0 + cc);
      adjtile[rr][cc]     = v.x;
      adjtile[rr][cc + 1] = v.y;
    }
    __syncthreads();

    if (m0 + 32 < NN)  // prefetch next K rows into L2/WGP$
      __builtin_prefetch(Qbh + (size_t)(m0 + 32 + col) * FOUT, 0, 1);

    // ---- S = Q K^T (two 16x16 tiles covering 32 keys)
    v8f s0 = {}, s1 = {};
#pragma unroll
    for (int kc = 0; kc < 4; ++kc) {
      v16bf bK = *(const v16bf*)(Qbh + (size_t)(m0 + col) * FOUT + kc * 32 + hf * 16);
      s0 = WMMA_BF16(aQ[kc], bK, s0);
    }
#pragma unroll
    for (int kc = 0; kc < 4; ++kc) {
      v16bf bK = *(const v16bf*)(Qbh + (size_t)(m0 + 16 + col) * FOUT + kc * 32 + hf * 16);
      s1 = WMMA_BF16(aQ[kc], bK, s1);
    }
    // ---- multiplicative adjacency + scale (pre-softmax, as in reference)
#pragma unroll
    for (int r = 0; r < 8; ++r) {
      int rr = r + 8 * hf;
      s0[r] *= scale * adjtile[rr][col];
      s1[r] *= scale * adjtile[rr][16 + col];
    }
    // ---- online softmax (row = query; reduce across the 16 lanes of a half)
    float p0[8], p1[8];
#pragma unroll
    for (int r = 0; r < 8; ++r) {
      float mx = fmaxf(s0[r], s1[r]);
      mx = fmaxf(mx, __shfl_xor(mx, 1, 32));
      mx = fmaxf(mx, __shfl_xor(mx, 2, 32));
      mx = fmaxf(mx, __shfl_xor(mx, 4, 32));
      mx = fmaxf(mx, __shfl_xor(mx, 8, 32));
      float mnew  = fmaxf(mrow[r], mx);
      float alpha = __expf(mrow[r] - mnew);
      mrow[r] = mnew;
      p0[r] = __expf(s0[r] - mnew);
      p1[r] = __expf(s1[r] - mnew);
      float rs = p0[r] + p1[r];
      rs += __shfl_xor(rs, 1, 32);
      rs += __shfl_xor(rs, 2, 32);
      rs += __shfl_xor(rs, 4, 32);
      rs += __shfl_xor(rs, 8, 32);
      lrow[r] = lrow[r] * alpha + rs;
#pragma unroll
      for (int c = 0; c < 8; ++c) O[c][r] *= alpha;
    }
    // ---- P: C-layout f32 -> A-layout bf16 via wave-private LDS (stride 40
    //      halves => the 16B read chunks hit disjoint bank quads).
#pragma unroll
    for (int r = 0; r < 8; ++r) {
      int rr = r + 8 * hf;
      pb[rr * 40 + col]      = (__bf16)p0[r];
      pb[rr * 40 + 16 + col] = (__bf16)p1[r];
    }
    asm volatile("" ::: "memory");   // keep LDS reads below the writes (LDS is in-order per wave)
    v8bf plo = *(const v8bf*)(pb + col * 40 + hf * 16);
    v8bf phi = *(const v8bf*)(pb + col * 40 + hf * 16 + 8);
    v16bf pA = __builtin_shufflevector(plo, phi, 0,1,2,3,4,5,6,7,8,9,10,11,12,13,14,15);
    // ---- O += P V  (V B-frags are contiguous 32B reads from transposed Qt)
#pragma unroll
    for (int c = 0; c < 8; ++c) {
      v16bf bV = *(const v16bf*)(Qth + (size_t)(c * 16 + col) * NN + m0 + hf * 16);
      O[c] = WMMA_BF16(pA, bV, O[c]);
    }
    __syncthreads();  // adjtile consumed; safe to restage next iteration
  }

  // ---- finalize (O /= l) and sum the 8 heads through LDS, store f32 out
  float invl[8];
#pragma unroll
  for (int r = 0; r < 8; ++r) invl[r] = 1.0f / lrow[r];

  for (int c = 0; c < 8; ++c) {
#pragma unroll
    for (int r = 0; r < 8; ++r)
      oredux[h][r + 8 * hf][col] = O[c][r] * invl[r];
    __syncthreads();
    {
      int row = threadIdx.x >> 4, cc = threadIdx.x & 15;
      float s = 0.0f;
#pragma unroll
      for (int w = 0; w < 8; ++w) s += oredux[w][row][cc];
      out[((size_t)b * NN + n0 + row) * FOUT + c * 16 + cc] = s;
    }
    __syncthreads();
  }
}

// ---------------------------------------------------------------------------
extern "C" void kernel_launch(void* const* d_in, const int* in_sizes, int n_in,
                              void* d_out, int out_size, void* d_ws, size_t ws_size,
                              hipStream_t stream) {
  (void)in_sizes; (void)n_in; (void)out_size; (void)ws_size;
  const float* X   = (const float*)d_in[0];   // [B,N,Fin]    f32
  const float* adj = (const float*)d_in[1];   // [B,N,N]      f32
  const float* W   = (const float*)d_in[2];   // [H,Fin,Fout] f32
  float* out = (float*)d_out;                 // [B,N,Fout]   f32

  __bf16* Qb  = (__bf16*)d_ws;                              // 16 MB [h,b,n,f]
  __bf16* Qt  = Qb  + (size_t)HN * BN * NN * FOUT;          // 16 MB [h,b,f,n]
  __bf16* Xbf = Qt  + (size_t)HN * BN * NN * FOUT;          //  4 MB [b,n,k]
  __bf16* Wt  = Xbf + (size_t)BN * NN * FIN;                // 0.5 MB [h,f,k]

  gat_cvt_kernel<<<dim3(3072), 256, 0, stream>>>(X, W, Xbf, Wt);

  dim3 g1(HN * BN, NN / 128);   // (64, 8)
  gat_proj_kernel<<<g1, 256, 0, stream>>>(Xbf, Wt, Qb, Qt);

  dim3 g2(BN, NN / 16);         // (8, 64)
  gat_attn_kernel<<<g2, 256, 0, stream>>>(adj, Qb, Qt, out);
}
